// FastRCNN_18090402251364
// MI455X (gfx1250) — matrix operand order repaired and checked by reading, not compiled, not run on hardware
//
#include <hip/hip_runtime.h>
#include <float.h>

typedef __attribute__((ext_vector_type(16))) _Float16 v16h;
typedef __attribute__((ext_vector_type(8)))  float    v8f;

#define NANCH   22200     // 12 * 37 * 50
#define NPAD    32768
#define PRE_NMS 6000
#define POST_NMS 300
#define FH 37
#define FW 50

// A-fragment K permutation: groups {0,2,1,3} of 8 so a lane's 16 values
// (ks = hi*8+0..7 and 16+hi*8+0..7) are one contiguous 32B LDS load.
__device__ __forceinline__ int kperm(int k) {
  int g = k >> 3;
  return ((((g & 1) << 1) | (g >> 1)) << 3) + (k & 7);
}

// ---------------------------------------------------------------------------
// WMMA 3x3 SAME conv, fused bias + optional ReLU.
// GEMM view: out[co][p] = sum_k w[co][k] * patch[k][p],  k = cin*9 + kh*3 + kw
// Block tile 64(Cout) x 128(pixels); 8 waves as 4(M) x 2(N);
// wave tile 16M x 64N = 4 WMMAs per K=32 chunk (one shared A fragment).
// ---------------------------------------------------------------------------
__global__ __launch_bounds__(256) void conv3x3_wmma(
    const float* __restrict__ x, const float* __restrict__ w,
    const float* __restrict__ bias, float* __restrict__ y,
    int Cin, int Cout, int H, int W, int relu)
{
  __shared__ alignas(64) _Float16 As[64][32];   // [co][k]
  __shared__ alignas(64) _Float16 Bs[128][32];  // [pixel][k]
  const int HW = H * W;
  const int K  = Cin * 9;
  const int p0 = blockIdx.x * 128;
  const int c0 = blockIdx.y * 64;
  const int tid  = threadIdx.x;
  const int lane = tid & 31;
  const int wave = tid >> 5;
  const int wm = wave >> 1;      // 0..3  (M sub-tile)
  const int wn = wave & 1;       // 0..1  (N sub-tile of 64 pixels)
  v8f acc0 = {}, acc1 = {}, acc2 = {}, acc3 = {};

  // Precompute this thread's 16 staging pixels (y,x) once (B k fixed = tid>>3,
  // n = (tid&7) + 8*j so 8 adjacent lanes stream contiguous x for coalescing).
  int py[16], px[16], pok[16];
  #pragma unroll
  for (int j = 0; j < 16; ++j) {
    int p = p0 + (tid & 7) + 8 * j;
    int ok = p < HW;
    int pp = ok ? p : 0;
    int yy = pp / W;
    py[j] = yy; px[j] = pp - yy * W; pok[j] = ok;
  }

  for (int k0 = 0; k0 < K; k0 += 32) {
    // ---- stage A (weights): 2048 elems, 8/thread, lanes contiguous in k ----
    {
      float tv[8];
      #pragma unroll
      for (int j = 0; j < 8; ++j) {
        int idx = tid + 256 * j;
        int k = idx & 31, m = idx >> 5;
        int kk = k0 + k;
        int pr = (c0 + m < Cout) & (kk < K);
        size_t a = pr ? ((size_t)(c0 + m) * K + kk) : (size_t)0;
        float t = w[a];
        tv[j] = pr ? t : 0.f;
      }
      #pragma unroll
      for (int j = 0; j < 8; ++j) {
        int idx = tid + 256 * j;
        int k = idx & 31, m = idx >> 5;
        As[m][k] = (_Float16)tv[j];
      }
    }
    // ---- stage B (im2col): 4096 elems, 16/thread, k fixed per thread ----
    {
      int kq = k0 + (tid >> 3);
      int kv = kq < K;
      int kc = kv ? kq : 0;
      int cin = kc / 9, r = kc - cin * 9;
      int kh = r / 3, kw = r - kh * 3;
      size_t cbase = (size_t)cin * HW;
      float tv[16];
      #pragma unroll
      for (int j = 0; j < 16; ++j) {
        int iy = py[j] + kh - 1;
        int ix = px[j] + kw - 1;
        int v = kv & pok[j] & (iy >= 0) & (iy < H) & (ix >= 0) & (ix < W);
        size_t a = v ? (cbase + (size_t)iy * W + ix) : (size_t)0;
        float t = x[a];
        tv[j] = v ? t : 0.f;
      }
      #pragma unroll
      for (int j = 0; j < 16; ++j)
        Bs[(tid & 7) + 8 * j][tid >> 3] = (_Float16)tv[j];
    }
    __syncthreads();

    const int hi   = lane >> 4;
    const int mrow = wm * 16 + (lane & 15);
    v16h a  = *(const v16h*)&As[mrow][hi * 16];
    const int nb = wn * 64 + (lane & 15);
    v16h b0 = *(const v16h*)&Bs[nb +  0][hi * 16];
    v16h b1 = *(const v16h*)&Bs[nb + 16][hi * 16];
    v16h b2 = *(const v16h*)&Bs[nb + 32][hi * 16];
    v16h b3 = *(const v16h*)&Bs[nb + 48][hi * 16];
    acc0 = __builtin_amdgcn_wmma_f32_16x16x32_f16(false, a, false, b0,
                                                  (short)0, acc0, false, false);
    acc1 = __builtin_amdgcn_wmma_f32_16x16x32_f16(false, a, false, b1,
                                                  (short)0, acc1, false, false);
    acc2 = __builtin_amdgcn_wmma_f32_16x16x32_f16(false, a, false, b2,
                                                  (short)0, acc2, false, false);
    acc3 = __builtin_amdgcn_wmma_f32_16x16x32_f16(false, a, false, b3,
                                                  (short)0, acc3, false, false);
    __syncthreads();
  }

  // C/D layout: lane n = lane&15; VGPR r -> M row r + 8*(lane>>4)
  const int hi = lane >> 4;
  #pragma unroll
  for (int r = 0; r < 8; ++r) {
    int m = c0 + wm * 16 + r + 8 * hi;
    if (m >= Cout) continue;
    float bs = bias[m];
    int na = p0 + wn * 64 + (lane & 15);
    float v0 = acc0[r] + bs, v1 = acc1[r] + bs;
    float v2 = acc2[r] + bs, v3 = acc3[r] + bs;
    if (relu) {
      v0 = fmaxf(v0, 0.f); v1 = fmaxf(v1, 0.f);
      v2 = fmaxf(v2, 0.f); v3 = fmaxf(v3, 0.f);
    }
    size_t row = (size_t)m * HW;
    if (na      < HW) y[row + na]      = v0;
    if (na + 16 < HW) y[row + na + 16] = v1;
    if (na + 32 < HW) y[row + na + 32] = v2;
    if (na + 48 < HW) y[row + na + 48] = v3;
  }
}

// ---------------------------------------------------------------------------
// WMMA GEMM: Y[M,N] = relu(X[M,K] @ Wt[K,N] + bias[N]); tile 64M x 128N
// ---------------------------------------------------------------------------
__global__ __launch_bounds__(256) void gemm_wmma(
    const float* __restrict__ X, const float* __restrict__ Wt,
    const float* __restrict__ bias, float* __restrict__ Y,
    int M, int N, int K, int relu)
{
  __shared__ alignas(64) _Float16 As[64][32];
  __shared__ alignas(64) _Float16 Bs[128][32];
  const int n0 = blockIdx.x * 128;
  const int m0 = blockIdx.y * 64;
  const int tid = threadIdx.x, lane = tid & 31, wave = tid >> 5;
  const int wm = wave >> 1, wn = wave & 1;
  v8f acc0 = {}, acc1 = {}, acc2 = {}, acc3 = {};

  for (int k0 = 0; k0 < K; k0 += 32) {
    if (k0 + 32 < K)  // pull the next weight panel toward L2
      __builtin_prefetch(&Wt[(size_t)(k0 + 32) * N + n0], 0, 1);
    // A panel: lanes contiguous in k
    {
      float tv[8];
      #pragma unroll
      for (int j = 0; j < 8; ++j) {
        int idx = tid + 256 * j;
        int k = idx & 31, m = idx >> 5;
        int mm = m0 + m, kk = k0 + k;
        int pr = (mm < M) & (kk < K);
        size_t a = pr ? ((size_t)mm * K + kk) : (size_t)0;
        float t = X[a];
        tv[j] = pr ? t : 0.f;
      }
      #pragma unroll
      for (int j = 0; j < 8; ++j) {
        int idx = tid + 256 * j;
        int k = idx & 31, m = idx >> 5;
        As[m][k] = (_Float16)tv[j];
      }
    }
    // B panel: lanes contiguous in n
    {
      float tv[16];
      #pragma unroll
      for (int j = 0; j < 16; ++j) {
        int idx = tid + 256 * j;
        int n = idx & 127, k = idx >> 7;
        int nn = n0 + n, kk = k0 + k;
        int pr = (nn < N) & (kk < K);
        size_t a = pr ? ((size_t)kk * N + nn) : (size_t)0;
        float t = Wt[a];
        tv[j] = pr ? t : 0.f;
      }
      #pragma unroll
      for (int j = 0; j < 16; ++j) {
        int idx = tid + 256 * j;
        Bs[idx & 127][idx >> 7] = (_Float16)tv[j];
      }
    }
    __syncthreads();
    const int hi = lane >> 4;
    v16h a  = *(const v16h*)&As[wm * 16 + (lane & 15)][hi * 16];
    const int nb = wn * 64 + (lane & 15);
    v16h b0 = *(const v16h*)&Bs[nb +  0][hi * 16];
    v16h b1 = *(const v16h*)&Bs[nb + 16][hi * 16];
    v16h b2 = *(const v16h*)&Bs[nb + 32][hi * 16];
    v16h b3 = *(const v16h*)&Bs[nb + 48][hi * 16];
    acc0 = __builtin_amdgcn_wmma_f32_16x16x32_f16(false, a, false, b0,
                                                  (short)0, acc0, false, false);
    acc1 = __builtin_amdgcn_wmma_f32_16x16x32_f16(false, a, false, b1,
                                                  (short)0, acc1, false, false);
    acc2 = __builtin_amdgcn_wmma_f32_16x16x32_f16(false, a, false, b2,
                                                  (short)0, acc2, false, false);
    acc3 = __builtin_amdgcn_wmma_f32_16x16x32_f16(false, a, false, b3,
                                                  (short)0, acc3, false, false);
    __syncthreads();
  }
  const int hi = lane >> 4;
  #pragma unroll
  for (int r = 0; r < 8; ++r) {
    int m = m0 + wm * 16 + r + 8 * hi;
    if (m >= M) continue;
    int na = n0 + wn * 64 + (lane & 15);
    size_t row = (size_t)m * N;
    #pragma unroll
    for (int q = 0; q < 4; ++q) {
      int n = na + 16 * q;
      if (n >= N) continue;
      float v = (q == 0 ? acc0[r] : q == 1 ? acc1[r] : q == 2 ? acc2[r]
                                                              : acc3[r]);
      v += bias[n];
      if (relu) v = fmaxf(v, 0.f);
      Y[row + n] = v;
    }
  }
}

// ---------------------------------------------------------------------------
// 2x2 stride-2 max pool (VALID)
// ---------------------------------------------------------------------------
__global__ void maxpool2(const float* __restrict__ x, float* __restrict__ y,
                         int C, int H, int W)
{
  int Ho = H >> 1, Wo = W >> 1;
  long long i = (long long)blockIdx.x * blockDim.x + threadIdx.x;
  long long total = (long long)C * Ho * Wo;
  if (i >= total) return;
  int c = (int)(i / ((long long)Ho * Wo));
  int rem = (int)(i - (long long)c * Ho * Wo);
  int yo = rem / Wo, xo = rem - yo * Wo;
  const float* p = x + ((size_t)c * H + 2 * yo) * W + 2 * xo;
  y[i] = fmaxf(fmaxf(p[0], p[1]), fmaxf(p[W], p[W + 1]));
}

// ---------------------------------------------------------------------------
// 1x1 conv (RPN score / bbox heads) — tiny, VALU is fine
// ---------------------------------------------------------------------------
__global__ void conv1x1(const float* __restrict__ x, const float* __restrict__ w,
                        const float* __restrict__ b, float* __restrict__ y,
                        int Cin, int Cout, int HW)
{
  int i = blockIdx.x * blockDim.x + threadIdx.x;
  if (i >= Cout * HW) return;
  int co = i / HW, p = i - co * HW;
  float acc = b[co];
  const float* wr = w + (size_t)co * Cin;
  for (int c = 0; c < Cin; ++c) acc += wr[c] * x[(size_t)c * HW + p];
  y[i] = acc;
}

// softmax over (bg,fg) channel pair -> foreground prob, (H,W,A) order
__global__ void rpn_fg_prob(const float* __restrict__ score,
                            float* __restrict__ sc, int HW)
{
  int i = blockIdx.x * blockDim.x + threadIdx.x;
  if (i >= HW * 12) return;
  int a = i % 12, p = i / 12;
  float s0 = score[(size_t)a * HW + p];
  float s1 = score[(size_t)(12 + a) * HW + p];
  sc[i] = 1.f / (1.f + expf(s0 - s1));   // == exp(s1)/(exp(s0)+exp(s1))
}

// ---------------------------------------------------------------------------
// Proposal layer: anchors + bbox decode + clip + min-size filter
// ---------------------------------------------------------------------------
__global__ void proposal_kernel(const float* __restrict__ bb,
                                const float* __restrict__ sc,
                                const float* __restrict__ im_info,
                                float* __restrict__ props4,
                                float* __restrict__ pscore,
                                int* __restrict__ pidx, int H, int W)
{
  int i = blockIdx.x * blockDim.x + threadIdx.x;
  if (i >= NPAD) return;
  if (i >= NANCH) { pscore[i] = -__builtin_inff(); pidx[i] = 0; return; }
  const int HW = H * W;
  int a = i % 12;
  int cell = i / 12;
  int gy = cell / W, gx = cell - gy * W;
  int ri = a >> 2, si = a & 3;
  float ratio = (ri == 0) ? 0.5f : (ri == 1 ? 1.f : 2.f);
  float scale = (float)(4 << si);
  float wsr = roundf(sqrtf(256.f / ratio));
  float hsr = roundf(wsr * ratio);
  float Wd = wsr * scale, Hd = hsr * scale;
  float sx = 16.f * gx, sy = 16.f * gy;
  float ax1 = 7.5f - 0.5f * (Wd - 1.f) + sx;
  float ay1 = 7.5f - 0.5f * (Hd - 1.f) + sy;
  float ax2 = 7.5f + 0.5f * (Wd - 1.f) + sx;
  float ay2 = 7.5f + 0.5f * (Hd - 1.f) + sy;
  float wA = ax2 - ax1 + 1.f, hA = ay2 - ay1 + 1.f;
  float cx = ax1 + 0.5f * wA, cy = ay1 + 0.5f * hA;
  int p = cell;
  float dx = bb[(size_t)(4 * a + 0) * HW + p];
  float dy = bb[(size_t)(4 * a + 1) * HW + p];
  float dw = bb[(size_t)(4 * a + 2) * HW + p];
  float dh = bb[(size_t)(4 * a + 3) * HW + p];
  float pcx = dx * wA + cx, pcy = dy * hA + cy;
  float pw = expf(dw) * wA, ph = expf(dh) * hA;
  float im_h = im_info[0], im_w = im_info[1], im_s = im_info[2];
  float x1 = fminf(fmaxf(pcx - 0.5f * pw, 0.f), im_w - 1.f);
  float y1 = fminf(fmaxf(pcy - 0.5f * ph, 0.f), im_h - 1.f);
  float x2 = fminf(fmaxf(pcx + 0.5f * pw, 0.f), im_w - 1.f);
  float y2 = fminf(fmaxf(pcy + 0.5f * ph, 0.f), im_h - 1.f);
  props4[(size_t)i * 4 + 0] = x1;
  props4[(size_t)i * 4 + 1] = y1;
  props4[(size_t)i * 4 + 2] = x2;
  props4[(size_t)i * 4 + 3] = y2;
  float ms = 16.f * im_s;
  bool ok = (x2 - x1 + 1.f >= ms) && (y2 - y1 + 1.f >= ms);
  pscore[i] = ok ? sc[i] : -__builtin_inff();
  pidx[i] = i;
}

// descending bitonic sort step over NPAD elements
__global__ void bitonic_step(float* __restrict__ s, int* __restrict__ idx,
                             int j, int k)
{
  int i = blockIdx.x * blockDim.x + threadIdx.x;
  int ixj = i ^ j;
  if (ixj <= i || ixj >= NPAD) return;
  bool up = (i & k) == 0;
  float si = s[i], sj = s[ixj];
  bool sw = up ? (si < sj) : (si > sj);
  if (sw) {
    s[i] = sj; s[ixj] = si;
    int t = idx[i]; idx[i] = idx[ixj]; idx[ixj] = t;
  }
}

__global__ void nms_prep(const float* __restrict__ props4,
                         const int* __restrict__ sidx,
                         float* __restrict__ sb, float* __restrict__ sarea)
{
  int j = blockIdx.x * blockDim.x + threadIdx.x;
  if (j >= PRE_NMS) return;
  int id = sidx[j];
  float x1 = props4[(size_t)id * 4 + 0];
  float y1 = props4[(size_t)id * 4 + 1];
  float x2 = props4[(size_t)id * 4 + 2];
  float y2 = props4[(size_t)id * 4 + 3];
  sb[(size_t)j * 4 + 0] = x1; sb[(size_t)j * 4 + 1] = y1;
  sb[(size_t)j * 4 + 2] = x2; sb[(size_t)j * 4 + 3] = y2;
  sarea[j] = (x2 - x1 + 1.f) * (y2 - y1 + 1.f);
}

// sequential NMS over score-sorted boxes; writes rois [0,x1,y1,x2,y2]*300
__global__ __launch_bounds__(256) void nms_kernel(const float* __restrict__ sb,
                                                  const float* __restrict__ sarea,
                                                  float* __restrict__ rois)
{
  __shared__ unsigned char sup[PRE_NMS];
  __shared__ int s_first, s_start;
  int tid = threadIdx.x;
  for (int j = tid; j < PRE_NMS; j += 256) sup[j] = 0;
  if (tid == 0) s_start = 0;
  __syncthreads();
  for (int t = 0; t < POST_NMS; ++t) {
    if (tid == 0) {
      int f = -1;
      for (int j = s_start; j < PRE_NMS; ++j)
        if (!sup[j]) { f = j; break; }
      s_first = f;
      if (f >= 0) s_start = f + 1;
    }
    __syncthreads();
    int i = s_first;
    if (i >= 0) {
      float x1 = sb[(size_t)i * 4], y1 = sb[(size_t)i * 4 + 1];
      float x2 = sb[(size_t)i * 4 + 2], y2 = sb[(size_t)i * 4 + 3];
      float ai = sarea[i];
      for (int j = tid; j < PRE_NMS; j += 256) {
        float iw = fmaxf(0.f, fminf(x2, sb[(size_t)j * 4 + 2]) -
                              fmaxf(x1, sb[(size_t)j * 4 + 0]) + 1.f);
        float ih = fmaxf(0.f, fminf(y2, sb[(size_t)j * 4 + 3]) -
                              fmaxf(y1, sb[(size_t)j * 4 + 1]) + 1.f);
        float inter = iw * ih;
        float iou = inter / (ai + sarea[j] - inter);
        if (iou > 0.7f) sup[j] = 1;
      }
    }
    __syncthreads();
    if (tid == 0) {
      int k = (i >= 0) ? i : 0;
      rois[(size_t)t * 5 + 0] = 0.f;
      rois[(size_t)t * 5 + 1] = sb[(size_t)k * 4 + 0];
      rois[(size_t)t * 5 + 2] = sb[(size_t)k * 4 + 1];
      rois[(size_t)t * 5 + 3] = sb[(size_t)k * 4 + 2];
      rois[(size_t)t * 5 + 4] = sb[(size_t)k * 4 + 3];
    }
    __syncthreads();
  }
}

// ---------------------------------------------------------------------------
// ROI max pool 7x7 (spatial scale 1/16, KH=10 / KW=12 window caps)
// output flat (roi, c*49 + ph*7 + pw)
// ---------------------------------------------------------------------------
__global__ void roi_pool_kernel(const float* __restrict__ feat,
                                const float* __restrict__ rois,
                                float* __restrict__ out, int C, int H, int W)
{
  int i = blockIdx.x * blockDim.x + threadIdx.x;
  int total = POST_NMS * C * 49;
  if (i >= total) return;
  int pw = i % 7; int t = i / 7;
  int ph = t % 7; t /= 7;
  int c = t % C; int r = t / C;
  const float* roi = rois + (size_t)r * 5;
  int x1 = (int)roundf(roi[1] * 0.0625f);
  int y1 = (int)roundf(roi[2] * 0.0625f);
  int x2 = (int)roundf(roi[3] * 0.0625f);
  int y2 = (int)roundf(roi[4] * 0.0625f);
  float roi_w = fmaxf((float)(x2 - x1 + 1), 1.f);
  float roi_h = fmaxf((float)(y2 - y1 + 1), 1.f);
  float bw = roi_w / 7.f, bh = roi_h / 7.f;
  int hs = min(max((int)floorf(ph * bh) + y1, 0), H);
  int he = min(max((int)ceilf((ph + 1) * bh) + y1, 0), H);
  he = min(he, hs + 10);                       // KH cap
  int wsx = min(max((int)floorf(pw * bw) + x1, 0), W);
  int we = min(max((int)ceilf((pw + 1) * bw) + x1, 0), W);
  we = min(we, wsx + 12);                      // KW cap
  float m = -FLT_MAX;
  bool any = false;
  for (int yy = hs; yy < he; ++yy)
    for (int xx = wsx; xx < we; ++xx) {
      any = true;
      m = fmaxf(m, feat[((size_t)c * H + yy) * W + xx]);
    }
  out[i] = any ? m : 0.f;
}

__global__ void softmax_rows(const float* __restrict__ x, float* __restrict__ y,
                             int M, int N)
{
  int r = blockIdx.x * blockDim.x + threadIdx.x;
  if (r >= M) return;
  const float* p = x + (size_t)r * N;
  float mx = -FLT_MAX;
  for (int j = 0; j < N; ++j) mx = fmaxf(mx, p[j]);
  float s = 0.f;
  for (int j = 0; j < N; ++j) s += expf(p[j] - mx);
  for (int j = 0; j < N; ++j) y[(size_t)r * N + j] = expf(p[j] - mx) / s;
}

// ---------------------------------------------------------------------------
extern "C" void kernel_launch(void* const* d_in, const int* in_sizes, int n_in,
                              void* d_out, int out_size, void* d_ws, size_t ws_size,
                              hipStream_t stream)
{
  (void)in_sizes; (void)n_in; (void)out_size; (void)ws_size;
  const float* im_data = (const float*)d_in[0];
  const float* im_info = (const float*)d_in[1];
  const float* vw[13]; const float* vb[13];
  for (int i = 0; i < 13; ++i) {
    vw[i] = (const float*)d_in[2 + i];
    vb[i] = (const float*)d_in[15 + i];
  }
  const float* rpn_w   = (const float*)d_in[28];
  const float* rpn_b   = (const float*)d_in[29];
  const float* score_w = (const float*)d_in[30];
  const float* score_b = (const float*)d_in[31];
  const float* bbox_w  = (const float*)d_in[32];
  const float* bbox_b  = (const float*)d_in[33];
  const float* fc6_w   = (const float*)d_in[34];
  const float* fc6_b   = (const float*)d_in[35];
  const float* fc7_w   = (const float*)d_in[36];
  const float* fc7_b   = (const float*)d_in[37];
  const float* cls_w   = (const float*)d_in[38];
  const float* cls_b   = (const float*)d_in[39];
  const float* reg_w   = (const float*)d_in[40];
  const float* reg_b   = (const float*)d_in[41];
  float* out = (float*)d_out;

  // workspace carve-out (floats)
  float* ws = (float*)d_ws;
  size_t o = 0;
  float* actA = ws + o;  o += 30720000;   // 64*600*800
  float* actB = ws + o;  o += 30720000;
  float* scoreBuf = ws + o; o += 24 * FH * FW;
  float* bboxBuf  = ws + o; o += 48 * FH * FW;
  float* scBuf    = ws + o; o += NANCH;
  float* props4   = ws + o; o += (size_t)NANCH * 4;
  float* pscore   = ws + o; o += NPAD;
  int*   pidx     = (int*)(ws + o); o += NPAD;
  float* sboxes   = ws + o; o += (size_t)PRE_NMS * 4;
  float* sarea    = ws + o; o += PRE_NMS;
  float* fcA      = ws + o; o += (size_t)POST_NMS * 4096;
  float* fcB      = ws + o; o += (size_t)POST_NMS * 4096;
  float* clsBuf   = ws + o; o += (size_t)POST_NMS * 21;

  float* rois_out = out + 6300 + 25200;   // after cls_prob and bbox_pred
  float* reg_out  = out + 6300;

  auto conv = [&](const float* x, const float* w, const float* b, float* y,
                  int Cin, int Cout, int H, int W) {
    dim3 g((unsigned)((H * W + 127) / 128), (unsigned)((Cout + 63) / 64));
    conv3x3_wmma<<<g, 256, 0, stream>>>(x, w, b, y, Cin, Cout, H, W, 1);
  };
  auto pool = [&](const float* x, float* y, int C, int H, int W) {
    long long total = (long long)C * (H >> 1) * (W >> 1);
    maxpool2<<<(unsigned)((total + 255) / 256), 256, 0, stream>>>(x, y, C, H, W);
  };
  auto gemm = [&](const float* X, const float* Wt, const float* b, float* Y,
                  int M, int N, int K, int relu) {
    dim3 g((unsigned)((N + 127) / 128), (unsigned)((M + 63) / 64));
    gemm_wmma<<<g, 256, 0, stream>>>(X, Wt, b, Y, M, N, K, relu);
  };

  // ---- VGG16 backbone ----
  conv(im_data, vw[0], vb[0], actA, 3, 64, 600, 800);
  conv(actA,    vw[1], vb[1], actB, 64, 64, 600, 800);
  pool(actB, actA, 64, 600, 800);                       // 300x400
  conv(actA, vw[2], vb[2], actB, 64, 128, 300, 400);
  conv(actB, vw[3], vb[3], actA, 128, 128, 300, 400);
  pool(actA, actB, 128, 300, 400);                      // 150x200
  conv(actB, vw[4], vb[4], actA, 128, 256, 150, 200);
  conv(actA, vw[5], vb[5], actB, 256, 256, 150, 200);
  conv(actB, vw[6], vb[6], actA, 256, 256, 150, 200);
  pool(actA, actB, 256, 150, 200);                      // 75x100
  conv(actB, vw[7], vb[7], actA, 256, 512, 75, 100);
  conv(actA, vw[8], vb[8], actB, 512, 512, 75, 100);
  conv(actB, vw[9], vb[9], actA, 512, 512, 75, 100);
  pool(actA, actB, 512, 75, 100);                       // 37x50
  conv(actB, vw[10], vb[10], actA, 512, 512, FH, FW);
  conv(actA, vw[11], vb[11], actB, 512, 512, FH, FW);
  conv(actB, vw[12], vb[12], actA, 512, 512, FH, FW);   // feats in actA

  // ---- RPN ----
  conv(actA, rpn_w, rpn_b, actB, 512, 512, FH, FW);     // rpn_conv1 (relu)
  const int HWf = FH * FW;
  conv1x1<<<(24 * HWf + 255) / 256, 256, 0, stream>>>(actB, score_w, score_b,
                                                      scoreBuf, 512, 24, HWf);
  conv1x1<<<(48 * HWf + 255) / 256, 256, 0, stream>>>(actB, bbox_w, bbox_b,
                                                      bboxBuf, 512, 48, HWf);
  rpn_fg_prob<<<(NANCH + 255) / 256, 256, 0, stream>>>(scoreBuf, scBuf, HWf);

  // ---- proposal layer ----
  proposal_kernel<<<NPAD / 256, 256, 0, stream>>>(bboxBuf, scBuf, im_info,
                                                  props4, pscore, pidx, FH, FW);
  for (int k = 2; k <= NPAD; k <<= 1)
    for (int j = k >> 1; j > 0; j >>= 1)
      bitonic_step<<<NPAD / 256, 256, 0, stream>>>(pscore, pidx, j, k);
  nms_prep<<<(PRE_NMS + 255) / 256, 256, 0, stream>>>(props4, pidx, sboxes, sarea);
  nms_kernel<<<1, 256, 0, stream>>>(sboxes, sarea, rois_out);

  // ---- ROI pool + head ----
  {
    int total = POST_NMS * 512 * 49;
    roi_pool_kernel<<<(total + 255) / 256, 256, 0, stream>>>(actA, rois_out,
                                                             actB, 512, FH, FW);
  }
  gemm(actB, fc6_w, fc6_b, fcA, POST_NMS, 4096, 512 * 49, 1);
  gemm(fcA,  fc7_w, fc7_b, fcB, POST_NMS, 4096, 4096, 1);
  gemm(fcB,  cls_w, cls_b, clsBuf, POST_NMS, 21, 4096, 0);
  gemm(fcB,  reg_w, reg_b, reg_out, POST_NMS, 84, 4096, 0);
  softmax_rows<<<(POST_NMS + 255) / 256, 256, 0, stream>>>(clsBuf, out,
                                                           POST_NMS, 21);
}
